// RPN_52742198395509
// MI455X (gfx1250) — matrix operand order repaired and checked by reading, not compiled, not run on hardware
//
#include <hip/hip_runtime.h>
#include <hip/hip_bf16.h>

typedef __attribute__((ext_vector_type(16))) _Float16 v16h;
typedef __attribute__((ext_vector_type(8)))  _Float16 v8h;
typedef __attribute__((ext_vector_type(8)))  float    v8f;
typedef __attribute__((ext_vector_type(4)))  int      v4i;

#define HF 100
#define WF 100
#define PPOS (HF*WF)         // 10000 spatial positions
#define CIN 256
#define KKTOT 2304           // 256*9
#define NANCH 90000
#define PRE_K 2000
#define POST_K 1000
#define NGT 20
#define BBOX_CLAMP 4.135166556742356f
#define NMS_THRESH 0.7f

static __device__ __forceinline__ int imin(int a, int b){ return a < b ? a : b; }

// ---- CDNA5 async global->LDS copy (ASYNCcnt path), guarded for portability ----
// Probe-derived signature: (v4i AS(1)*, v4i AS(3)*, imm int offset, imm int cpol)
typedef __attribute__((address_space(1))) v4i as1_v4i;
typedef __attribute__((address_space(3))) v4i as3_v4i;

__device__ __forceinline__ void async_copy_b128(void* lds_dst, const void* gsrc) {
#if defined(__has_builtin) && __has_builtin(__builtin_amdgcn_global_load_async_to_lds_b128)
  __builtin_amdgcn_global_load_async_to_lds_b128(
      (as1_v4i*)(unsigned long long)(size_t)gsrc,
      (as3_v4i*)(unsigned int)(size_t)lds_dst, 0, 0);
#else
  *(v8h*)lds_dst = *(const v8h*)gsrc;
#endif
}

__device__ __forceinline__ void async_wait0() {
#if defined(__has_builtin) && __has_builtin(__builtin_amdgcn_s_wait_asynccnt)
  __builtin_amdgcn_s_wait_asynccnt(0);
#else
  asm volatile("s_wait_asynccnt 0x0" ::: "memory");
#endif
}

// ---------------------------------------------------------------- prep
// Materialize im2col: A[p][kk] (f16), p in [0,10000), kk = c*9+ky*3+kx in [0,2304)
__global__ void k_im2col(const float* __restrict__ feat, _Float16* __restrict__ aim) {
  int i = blockIdx.x * blockDim.x + threadIdx.x;
  if (i >= PPOS * KKTOT) return;
  int p = i / KKTOT;
  int kk = i - p * KKTOT;
  int c = kk / 9;
  int rem = kk - c * 9;
  int ky = rem / 3, kx = rem - ky * 3;
  int y = p / WF, x = p - y * WF;
  int yy = y + ky - 1, xx = x + kx - 1;
  float v = 0.0f;
  if (yy >= 0 && yy < HF && xx >= 0 && xx < WF)
    v = feat[(c * HF + yy) * WF + xx];
  aim[i] = (_Float16)v;
}

__global__ void k_prep_w(const float* __restrict__ w, _Float16* __restrict__ wh, int n) {
  int i = blockIdx.x * blockDim.x + threadIdx.x;
  if (i < n) wh[i] = (_Float16)w[i];
}

// ------------------------------------------------ 3x3 conv as GEMM + WMMA
// M = spatial (10000), N = out channels (256), K = 2304.
// Block = 128 (4 waves); wave w computes M-tile blockIdx.x x N-tiles [4w..4w+3].
// A tile staged to LDS via async b128 copies; fragments are ds_load_b128 pairs.
__global__ __launch_bounds__(128)
void k_conv3x3_wmma(const _Float16* __restrict__ aim,   // [p][k] = [10000][2304]
                    const _Float16* __restrict__ wB,    // [n][k] = [256][2304]
                    const float* __restrict__ bias,
                    float* __restrict__ out)            // [ch][p] = [256][10000]
{
  const int m0   = blockIdx.x;                 // 0..624
  const int wave = threadIdx.x >> 5;
  const int lane = threadIdx.x & 31;
  const int r    = lane & 15;
  const int hi   = lane >> 4;
  const int col  = lane & 15;

  __shared__ __attribute__((aligned(16))) _Float16 lds_a[16 * 32];

  v8f acc0 = {}, acc1 = {}, acc2 = {}, acc3 = {};

  // per-wave N columns for the 4 tiles
  const int ch0 = (wave * 4 + 0) * 16 + col;
  const int ch1 = (wave * 4 + 1) * 16 + col;
  const int ch2 = (wave * 4 + 2) * 16 + col;
  const int ch3 = (wave * 4 + 3) * 16 + col;
  const _Float16* b0 = wB + (size_t)ch0 * KKTOT + hi * 16;
  const _Float16* b1 = wB + (size_t)ch1 * KKTOT + hi * 16;
  const _Float16* b2 = wB + (size_t)ch2 * KKTOT + hi * 16;
  const _Float16* b3 = wB + (size_t)ch3 * KKTOT + hi * 16;

  // A staging: thread t (0..63) copies halves [t*8, t*8+8) of the 16x32 tile
  const int arow = threadIdx.x >> 2;                 // t/4 = row
  const int akl  = (threadIdx.x & 3) * 8;            // (t%4)*8 = k offset
  const _Float16* asrc0 = aim + (size_t)(m0 * 16 + arow) * KKTOT + akl;

  for (int kb = 0; kb < KKTOT / 32; ++kb) {
    __syncthreads();                                  // protect prior reads
    if (threadIdx.x < 64)
      async_copy_b128(&lds_a[threadIdx.x * 8], asrc0 + kb * 32);
    async_wait0();
    __syncthreads();

    // A fragment: lanes 0-15 hold K 0..15, lanes 16-31 hold K 16..31
    v8h alo = *(const v8h*)&lds_a[r * 32 + hi * 16];
    v8h ahi = *(const v8h*)&lds_a[r * 32 + hi * 16 + 8];
    v16h a;
#pragma unroll
    for (int t = 0; t < 8; ++t) { a[t] = alo[t]; a[t + 8] = ahi[t]; }

    // B fragments: lane = N column, halves = K = hi*16 + h (two b128 loads each)
    const _Float16* p0 = b0 + kb * 32;
    const _Float16* p1 = b1 + kb * 32;
    const _Float16* p2 = b2 + kb * 32;
    const _Float16* p3 = b3 + kb * 32;
    __builtin_prefetch(p0 + 32, 0, 0);
    v8h q; v16h bf;

    q = *(const v8h*)(p0);
#pragma unroll
    for (int t = 0; t < 8; ++t) bf[t] = q[t];
    q = *(const v8h*)(p0 + 8);
#pragma unroll
    for (int t = 0; t < 8; ++t) bf[t + 8] = q[t];
    acc0 = __builtin_amdgcn_wmma_f32_16x16x32_f16(false, a, false, bf, (short)0, acc0, false, false);

    q = *(const v8h*)(p1);
#pragma unroll
    for (int t = 0; t < 8; ++t) bf[t] = q[t];
    q = *(const v8h*)(p1 + 8);
#pragma unroll
    for (int t = 0; t < 8; ++t) bf[t + 8] = q[t];
    acc1 = __builtin_amdgcn_wmma_f32_16x16x32_f16(false, a, false, bf, (short)0, acc1, false, false);

    q = *(const v8h*)(p2);
#pragma unroll
    for (int t = 0; t < 8; ++t) bf[t] = q[t];
    q = *(const v8h*)(p2 + 8);
#pragma unroll
    for (int t = 0; t < 8; ++t) bf[t + 8] = q[t];
    acc2 = __builtin_amdgcn_wmma_f32_16x16x32_f16(false, a, false, bf, (short)0, acc2, false, false);

    q = *(const v8h*)(p3);
#pragma unroll
    for (int t = 0; t < 8; ++t) bf[t] = q[t];
    q = *(const v8h*)(p3 + 8);
#pragma unroll
    for (int t = 0; t < 8; ++t) bf[t + 8] = q[t];
    acc3 = __builtin_amdgcn_wmma_f32_16x16x32_f16(false, a, false, bf, (short)0, acc3, false, false);
  }

  // D layout: VGPR v -> row v + 8*hi, lane -> column
  float bv0 = bias[ch0], bv1 = bias[ch1], bv2 = bias[ch2], bv3 = bias[ch3];
#pragma unroll
  for (int v = 0; v < 8; ++v) {
    int prow = m0 * 16 + hi * 8 + v;
    float t0 = acc0[v] + bv0; out[(size_t)ch0 * PPOS + prow] = t0 > 0.0f ? t0 : 0.0f;
    float t1 = acc1[v] + bv1; out[(size_t)ch1 * PPOS + prow] = t1 > 0.0f ? t1 : 0.0f;
    float t2 = acc2[v] + bv2; out[(size_t)ch2 * PPOS + prow] = t2 > 0.0f ? t2 : 0.0f;
    float t3 = acc3[v] + bv3; out[(size_t)ch3 * PPOS + prow] = t3 > 0.0f ? t3 : 0.0f;
  }
}

// ---------------------------------------------------------------- 1x1 heads
__global__ void k_heads(const float* __restrict__ conv,
                        const float* __restrict__ w_reg, const float* __restrict__ b_reg,
                        const float* __restrict__ w_cls, const float* __restrict__ b_cls,
                        float* __restrict__ reg, float* __restrict__ cls) {
  int tid = blockIdx.x * blockDim.x + threadIdx.x;
  if (tid >= PPOS * 45) return;
  int p = tid / 45, o = tid - p * 45;
  const float* wrow; float s;
  if (o < 36) { wrow = w_reg + o * CIN; s = b_reg[o]; }
  else        { wrow = w_cls + (o - 36) * CIN; s = b_cls[o - 36]; }
  for (int c = 0; c < CIN; ++c) s += wrow[c] * conv[(size_t)c * PPOS + p];
  if (o < 36) reg[(p * 9 + (o >> 2)) * 4 + (o & 3)] = s;
  else        cls[p * 9 + (o - 36)] = s;
}

// ----------------------------------------------------- anchors + deltas + clip
__global__ void k_proposals(const float* __restrict__ reg,
                            float* __restrict__ anchors, float* __restrict__ props) {
  int i = blockIdx.x * blockDim.x + threadIdx.x;
  if (i >= NANCH) return;
  int p = i / 9, a = i - p * 9;
  int y = p / WF, x = p - y * WF;
  const float ratios[3] = {0.5f, 1.0f, 2.0f};
  const float scales[3] = {128.0f, 256.0f, 512.0f};
  float hr = sqrtf(ratios[a / 3]);
  float wr = 1.0f / hr;
  float ws = wr * scales[a % 3], hs = hr * scales[a % 3];
  float acx = (float)x * 8.0f, acy = (float)y * 8.0f;
  anchors[i * 4 + 0] = acx - 0.5f * ws;
  anchors[i * 4 + 1] = acy - 0.5f * hs;
  anchors[i * 4 + 2] = acx + 0.5f * ws;
  anchors[i * 4 + 3] = acy + 0.5f * hs;
  float dx = reg[i * 4 + 0], dy = reg[i * 4 + 1];
  float dw = fminf(reg[i * 4 + 2], BBOX_CLAMP);
  float dh = fminf(reg[i * 4 + 3], BBOX_CLAMP);
  float cx = dx * ws + acx, cy = dy * hs + acy;
  float w = expf(dw) * ws, h = expf(dh) * hs;
  props[i * 4 + 0] = fminf(fmaxf(cx - 0.5f * w, 0.0f), 800.0f);
  props[i * 4 + 1] = fminf(fmaxf(cy - 0.5f * h, 0.0f), 800.0f);
  props[i * 4 + 2] = fminf(fmaxf(cx + 0.5f * w, 0.0f), 800.0f);
  props[i * 4 + 3] = fminf(fmaxf(cy + 0.5f * h, 0.0f), 800.0f);
}

// --------------------------------- exact top-k via rank selection (stable ties)
__global__ void k_rank_select(const float* __restrict__ vals, int N, int K,
                              int* __restrict__ out_idx, float* __restrict__ out_val) {
  __shared__ float sv[256];
  int tid = blockIdx.x * 256 + threadIdx.x;
  float myv = (tid < N) ? vals[tid] : 0.0f;
  int rank = 0;
  for (int base = 0; base < N; base += 256) {
    int j = base + threadIdx.x;
    sv[threadIdx.x] = (j < N) ? vals[j] : -__builtin_inff();
    __syncthreads();
    int lim = imin(256, N - base);
    for (int t = 0; t < lim; ++t) {
      float v = sv[t];
      int j2 = base + t;
      if (tid < N && (v > myv || (v == myv && j2 < tid))) ++rank;
    }
    __syncthreads();
  }
  if (tid < N && rank < K) { out_idx[rank] = tid; out_val[rank] = myv; }
}

__global__ void k_gather_pre(const int* __restrict__ sel_idx, const float* __restrict__ sel_val,
                             const float* __restrict__ props,
                             float* __restrict__ boxes, float* __restrict__ probs) {
  int t = blockIdx.x * blockDim.x + threadIdx.x;
  if (t >= PRE_K) return;
  int i = sel_idx[t];
  float x0 = props[i * 4 + 0], y0 = props[i * 4 + 1];
  float x1 = props[i * 4 + 2], y1 = props[i * 4 + 3];
  boxes[t * 4 + 0] = x0; boxes[t * 4 + 1] = y0;
  boxes[t * 4 + 2] = x1; boxes[t * 4 + 3] = y1;
  float pr = 1.0f / (1.0f + expf(-sel_val[t]));
  if (!((x1 - x0 >= 16.0f) && (y1 - y0 >= 16.0f))) pr = -__builtin_inff();
  probs[t] = pr;
}

__global__ void k_gather_sorted(const int* __restrict__ ord_idx, const float* __restrict__ ord_val,
                                const float* __restrict__ boxes,
                                float* __restrict__ sboxes, float* __restrict__ sprobs) {
  int t = blockIdx.x * blockDim.x + threadIdx.x;
  if (t >= PRE_K) return;
  int j = ord_idx[t];
  sboxes[t * 4 + 0] = boxes[j * 4 + 0];
  sboxes[t * 4 + 1] = boxes[j * 4 + 1];
  sboxes[t * 4 + 2] = boxes[j * 4 + 2];
  sboxes[t * 4 + 3] = boxes[j * 4 + 3];
  sprobs[t] = ord_val[t];
}

// ---------------------------------------------------------------- NMS (1 block)
__global__ __launch_bounds__(1024)
void k_nms(const float* __restrict__ sboxes, const float* __restrict__ sprobs,
           float* __restrict__ keptv) {
  __shared__ float bx[PRE_K * 4];
  __shared__ unsigned char keep[PRE_K];
  for (int t = threadIdx.x; t < PRE_K * 4; t += 1024) bx[t] = sboxes[t];
  for (int t = threadIdx.x; t < PRE_K; t += 1024) keep[t] = 1;
  __syncthreads();
  for (int i = 0; i < PRE_K; ++i) {
    if (keep[i]) {
      float ax0 = bx[i * 4 + 0], ay0 = bx[i * 4 + 1];
      float ax1 = bx[i * 4 + 2], ay1 = bx[i * 4 + 3];
      float aarea = (ax1 - ax0) * (ay1 - ay0);
      for (int j = i + 1 + threadIdx.x; j < PRE_K; j += 1024) {
        if (!keep[j]) continue;
        float bx0 = bx[j * 4 + 0], by0 = bx[j * 4 + 1];
        float bx1 = bx[j * 4 + 2], by1 = bx[j * 4 + 3];
        float iw = fminf(ax1, bx1) - fmaxf(ax0, bx0);
        float ih = fminf(ay1, by1) - fmaxf(ay0, by0);
        iw = fmaxf(iw, 0.0f); ih = fmaxf(ih, 0.0f);
        float inter = iw * ih;
        float barea = (bx1 - bx0) * (by1 - by0);
        float iou = inter / (aarea + barea - inter);
        if (iou > NMS_THRESH) keep[j] = 0;
      }
    }
    __syncthreads();
  }
  for (int t = threadIdx.x; t < PRE_K; t += 1024)
    keptv[t] = keep[t] ? sprobs[t] : -__builtin_inff();
}

__global__ void k_gather_final(const int* __restrict__ fin_idx, const float* __restrict__ fin_val,
                               const float* __restrict__ sboxes, float* __restrict__ out) {
  int t = blockIdx.x * blockDim.x + threadIdx.x;
  if (t >= POST_K) return;
  int j = fin_idx[t];
  out[t * 4 + 0] = sboxes[j * 4 + 0];
  out[t * 4 + 1] = sboxes[j * 4 + 1];
  out[t * 4 + 2] = sboxes[j * 4 + 2];
  out[t * 4 + 3] = sboxes[j * 4 + 3];
  out[4000 + t] = fin_val[t];
}

// ---------------------------------------------------------------- losses
__device__ __forceinline__ float iou_one(float ax0, float ay0, float ax1, float ay1,
                                         float gx0, float gy0, float gx1, float gy1) {
  float iw = fmaxf(fminf(ax1, gx1) - fmaxf(ax0, gx0), 0.0f);
  float ih = fmaxf(fminf(ay1, gy1) - fmaxf(ay0, gy0), 0.0f);
  float inter = iw * ih;
  float aa = (ax1 - ax0) * (ay1 - ay0);
  float ab = (gx1 - gx0) * (gy1 - gy0);
  return inter / (aa + ab - inter);
}

__global__ void k_init(int* npos, int* nsamp, float* sums) {
  npos[0] = 0; nsamp[0] = 0; sums[0] = 0.0f; sums[1] = 0.0f;
}

__global__ void k_best_per_gt(const float* __restrict__ anchors, const float* __restrict__ gt,
                              float* __restrict__ best) {
  int g = blockIdx.x;
  float gx0 = gt[g * 4], gy0 = gt[g * 4 + 1], gx1 = gt[g * 4 + 2], gy1 = gt[g * 4 + 3];
  float m = -__builtin_inff();
  for (int i = threadIdx.x; i < NANCH; i += 256) {
    m = fmaxf(m, iou_one(anchors[i * 4], anchors[i * 4 + 1], anchors[i * 4 + 2], anchors[i * 4 + 3],
                         gx0, gy0, gx1, gy1));
  }
  __shared__ float red[256];
  red[threadIdx.x] = m;
  __syncthreads();
  for (int s = 128; s > 0; s >>= 1) {
    if (threadIdx.x < s) red[threadIdx.x] = fmaxf(red[threadIdx.x], red[threadIdx.x + s]);
    __syncthreads();
  }
  if (threadIdx.x == 0) best[g] = red[0];
}

__device__ __forceinline__ float hash_u(unsigned int i) {
  i ^= i >> 16; i *= 0x7feb352du; i ^= i >> 15; i *= 0x846ca68bu; i ^= i >> 16;
  return (float)(i >> 8) * (1.0f / 16777216.0f);
}

__global__ void k_labels(const float* __restrict__ anchors, const float* __restrict__ gt,
                         const float* __restrict__ best,
                         float* __restrict__ labels, float* __restrict__ tgtd,
                         float* __restrict__ uarr) {
  __shared__ float sgt[NGT * 4];
  __shared__ float sbest[NGT];
  if (threadIdx.x < NGT * 4) sgt[threadIdx.x] = gt[threadIdx.x];
  if (threadIdx.x < NGT) sbest[threadIdx.x] = best[threadIdx.x];
  __syncthreads();
  int i = blockIdx.x * blockDim.x + threadIdx.x;
  if (i >= NANCH) return;
  float ax0 = anchors[i * 4], ay0 = anchors[i * 4 + 1];
  float ax1 = anchors[i * 4 + 2], ay1 = anchors[i * 4 + 3];
  float maxi = -__builtin_inff(); int arg = 0; bool force = false;
  for (int g = 0; g < NGT; ++g) {
    float v = iou_one(ax0, ay0, ax1, ay1,
                      sgt[g * 4], sgt[g * 4 + 1], sgt[g * 4 + 2], sgt[g * 4 + 3]);
    if (v > maxi) { maxi = v; arg = g; }
    if (v == sbest[g]) force = true;
  }
  float lab = (maxi >= 0.7f) ? 1.0f : ((maxi < 0.3f) ? 0.0f : -1.0f);
  if (force) lab = 1.0f;
  labels[i] = lab;
  float aw = ax1 - ax0, ah = ay1 - ay0;
  float acx = ax0 + 0.5f * aw, acy = ay0 + 0.5f * ah;
  float gx0 = sgt[arg * 4], gy0 = sgt[arg * 4 + 1];
  float gx1 = sgt[arg * 4 + 2], gy1 = sgt[arg * 4 + 3];
  float gw = gx1 - gx0, gh = gy1 - gy0;
  float gcx = gx0 + 0.5f * gw, gcy = gy0 + 0.5f * gh;
  tgtd[i * 4 + 0] = (gcx - acx) / aw;
  tgtd[i * 4 + 1] = (gcy - acy) / ah;
  tgtd[i * 4 + 2] = logf(gw / aw);
  tgtd[i * 4 + 3] = logf(gh / ah);
  uarr[i] = hash_u((unsigned)i);   // stand-in for jax threefry uniforms
}

__global__ void k_count_pos(const float* __restrict__ labels, int* __restrict__ npos) {
  int i = blockIdx.x * blockDim.x + threadIdx.x;
  if (i < NANCH && labels[i] == 1.0f) atomicAdd(npos, 1);
}

__global__ void k_sample(const float* __restrict__ labels, const float* __restrict__ u,
                         const int* __restrict__ npos_total,
                         unsigned char* __restrict__ pos_mask,
                         unsigned char* __restrict__ samp_mask,
                         int* __restrict__ n_samp) {
  __shared__ float su[256];
  __shared__ float sl[256];
  int tid = blockIdx.x * 256 + threadIdx.x;
  float myu = (tid < NANCH) ? u[tid] : 0.0f;
  float myl = (tid < NANCH) ? labels[tid] : -1.0f;
  int rp = 0, rn = 0;
  for (int base = 0; base < NANCH; base += 256) {
    int j = base + threadIdx.x;
    su[threadIdx.x] = (j < NANCH) ? u[j] : -__builtin_inff();
    sl[threadIdx.x] = (j < NANCH) ? labels[j] : -1.0f;
    __syncthreads();
    int lim = imin(256, NANCH - base);
    for (int t = 0; t < lim; ++t) {
      float uj = su[t]; float lj = sl[t]; int j2 = base + t;
      bool gr = (uj > myu) || (uj == myu && j2 < tid);
      if (gr && lj == 1.0f) ++rp;
      if (gr && lj == 0.0f) ++rn;
    }
    __syncthreads();
  }
  if (tid >= NANCH) return;
  int npc = imin(npos_total[0], 128);
  bool pos = (myl == 1.0f) && (rp < 128);
  bool neg = (myl == 0.0f) && (rn < 256 - npc);
  pos_mask[tid] = pos ? 1 : 0;
  samp_mask[tid] = (pos || neg) ? 1 : 0;
  if (pos || neg) atomicAdd(n_samp, 1);
}

__global__ void k_losses(const float* __restrict__ cls, const float* __restrict__ labels,
                         const float* __restrict__ reg, const float* __restrict__ tgtd,
                         const unsigned char* __restrict__ pos_mask,
                         const unsigned char* __restrict__ samp_mask,
                         float* __restrict__ sums) {
  int i = blockIdx.x * blockDim.x + threadIdx.x;
  if (i >= NANCH) return;
  if (samp_mask[i]) {
    float c = cls[i], lab = labels[i];
    float bce = fmaxf(c, 0.0f) - c * lab + log1pf(expf(-fabsf(c)));
    atomicAdd(&sums[0], bce);
  }
  if (pos_mask[i]) {
    const float beta = 1.0f / 9.0f;
    float s = 0.0f;
    for (int k = 0; k < 4; ++k) {
      float d = reg[i * 4 + k] - tgtd[i * 4 + k];
      float ad = fabsf(d);
      s += (ad < beta) ? 0.5f * d * d / beta : ad - 0.5f * beta;
    }
    atomicAdd(&sums[1], s);
  }
}

__global__ void k_finalize(const float* __restrict__ sums, const int* __restrict__ n_samp,
                           float* __restrict__ out) {
  float ns = (float)n_samp[0];
  out[5000] = sums[0] / ns;
  out[5001] = sums[1] / ns;
}

// ---------------------------------------------------------------- launcher
extern "C" void kernel_launch(void* const* d_in, const int* in_sizes, int n_in,
                              void* d_out, int out_size, void* d_ws, size_t ws_size,
                              hipStream_t stream) {
  const float* feat  = (const float*)d_in[1];
  const float* gt    = (const float*)d_in[2];
  const float* w_rpn = (const float*)d_in[3];
  const float* b_rpn = (const float*)d_in[4];
  const float* w_reg = (const float*)d_in[5];
  const float* b_reg = (const float*)d_in[6];
  const float* w_cls = (const float*)d_in[7];
  const float* b_cls = (const float*)d_in[8];
  float* out = (float*)d_out;

  char* ws = (char*)d_ws;
  size_t off = 0;
  auto take = [&](size_t bytes) -> void* {
    void* p = ws + off;
    off = (off + bytes + 255) & ~(size_t)255;
    return p;
  };
  _Float16* aim  = (_Float16*)take((size_t)PPOS * KKTOT * 2);   // im2col, ~44MB
  _Float16* wBh  = (_Float16*)take((size_t)CIN * KKTOT * 2);
  float* conv    = (float*)take((size_t)CIN * PPOS * 4);
  float* reg     = (float*)take((size_t)NANCH * 4 * 4);
  float* cls     = (float*)take((size_t)NANCH * 4);
  float* anchors = (float*)take((size_t)NANCH * 4 * 4);
  float* props   = (float*)take((size_t)NANCH * 4 * 4);
  int*   sel_idx = (int*)take(PRE_K * 4);
  float* sel_val = (float*)take(PRE_K * 4);
  float* boxes2k = (float*)take(PRE_K * 4 * 4);
  float* probs2k = (float*)take(PRE_K * 4);
  int*   ord_idx = (int*)take(PRE_K * 4);
  float* ord_val = (float*)take(PRE_K * 4);
  float* sboxes  = (float*)take(PRE_K * 4 * 4);
  float* sprobs  = (float*)take(PRE_K * 4);
  float* keptv   = (float*)take(PRE_K * 4);
  int*   fin_idx = (int*)take(POST_K * 4);
  float* fin_val = (float*)take(POST_K * 4);
  float* labels  = (float*)take((size_t)NANCH * 4);
  float* tgtd    = (float*)take((size_t)NANCH * 4 * 4);
  float* uarr    = (float*)take((size_t)NANCH * 4);
  float* best    = (float*)take(NGT * 4);
  int*   npos    = (int*)take(4);
  int*   nsamp   = (int*)take(4);
  float* sums    = (float*)take(2 * 4);
  unsigned char* posm  = (unsigned char*)take(NANCH);
  unsigned char* sampm = (unsigned char*)take(NANCH);

  k_init<<<1, 1, 0, stream>>>(npos, nsamp, sums);

  int nim = PPOS * KKTOT;
  k_im2col<<<(nim + 255) / 256, 256, 0, stream>>>(feat, aim);
  int nwe = CIN * KKTOT;
  k_prep_w<<<(nwe + 255) / 256, 256, 0, stream>>>(w_rpn, wBh, nwe);

  k_conv3x3_wmma<<<dim3(PPOS / 16), 128, 0, stream>>>(aim, wBh, b_rpn, conv);

  k_heads<<<(PPOS * 45 + 255) / 256, 256, 0, stream>>>(conv, w_reg, b_reg, w_cls, b_cls, reg, cls);
  k_proposals<<<(NANCH + 255) / 256, 256, 0, stream>>>(reg, anchors, props);

  // filter_proposals: top-2000, sigmoid+minsize, stable sort, NMS, top-1000
  k_rank_select<<<(NANCH + 255) / 256, 256, 0, stream>>>(cls, NANCH, PRE_K, sel_idx, sel_val);
  k_gather_pre<<<(PRE_K + 255) / 256, 256, 0, stream>>>(sel_idx, sel_val, props, boxes2k, probs2k);
  k_rank_select<<<(PRE_K + 255) / 256, 256, 0, stream>>>(probs2k, PRE_K, PRE_K, ord_idx, ord_val);
  k_gather_sorted<<<(PRE_K + 255) / 256, 256, 0, stream>>>(ord_idx, ord_val, boxes2k, sboxes, sprobs);
  k_nms<<<1, 1024, 0, stream>>>(sboxes, sprobs, keptv);
  k_rank_select<<<(PRE_K + 255) / 256, 256, 0, stream>>>(keptv, PRE_K, POST_K, fin_idx, fin_val);
  k_gather_final<<<(POST_K + 255) / 256, 256, 0, stream>>>(fin_idx, fin_val, sboxes, out);

  // losses
  k_best_per_gt<<<NGT, 256, 0, stream>>>(anchors, gt, best);
  k_labels<<<(NANCH + 255) / 256, 256, 0, stream>>>(anchors, gt, best, labels, tgtd, uarr);
  k_count_pos<<<(NANCH + 255) / 256, 256, 0, stream>>>(labels, npos);
  k_sample<<<(NANCH + 255) / 256, 256, 0, stream>>>(labels, uarr, npos, posm, sampm, nsamp);
  k_losses<<<(NANCH + 255) / 256, 256, 0, stream>>>(cls, labels, reg, tgtd, posm, sampm, sums);
  k_finalize<<<1, 1, 0, stream>>>(sums, nsamp, out);
}
